// MultiHeadAttention_53506702573771
// MI455X (gfx1250) — matrix operand order repaired and checked by reading, not compiled, not run on hardware
//
#include <hip/hip_runtime.h>
#include <hip/hip_bf16.h>

// GAT forward: h = x@W^T ; e = lrelu(e_src_i + e_dst_j) masked ; softmax_j ; out = alpha @ h
// N=3072, IN_F=512, H=8, D=64.  bf16 WMMA (v_wmma_f32_16x16x32_bf16), f32 accumulate.
// adj tiles staged to LDS by the Tensor Data Mover (tensor_load_to_lds / TENSORcnt).

typedef __attribute__((ext_vector_type(16))) __bf16 v16bf;
typedef __attribute__((ext_vector_type(8)))  __bf16 v8bf;
typedef __attribute__((ext_vector_type(8)))  float  v8f;
typedef __attribute__((ext_vector_type(4)))  unsigned int v4u;
typedef __attribute__((ext_vector_type(8)))  int v8i;
typedef __attribute__((ext_vector_type(4)))  int v4i;

#define NN    3072   // nodes
#define INF_  512    // input features
#define HD    512    // H*D output features
#define NHEAD 8
#define DHEAD 64

// ---------------------------------------------------------------------------
// Fragment loaders following CDNA5 ISA 7.12.2 VGPR layouts (wave32).
__device__ inline v16bf load_frag_a_rowmajor(const __bf16* __restrict__ src, int ld,
                                             int row0, int kb) {
    const int lane = threadIdx.x & 31;
    const int m    = lane & 15;
    const int half = lane >> 4;
    const __bf16* base = src + (size_t)(row0 + m) * ld + kb + 8 * half;
    v8bf lo = *(const v8bf*)(base);
    v8bf hi = *(const v8bf*)(base + 16);
    v16bf a;
#pragma unroll
    for (int i = 0; i < 8; ++i) { a[i] = lo[i]; a[i + 8] = hi[i]; }
    return a;
}

__device__ inline v16bf load_frag_b_kcontig(const __bf16* __restrict__ src, size_t ld,
                                            int col0, int kb) {
    const int lane  = threadIdx.x & 31;
    const int n     = lane & 15;
    const int khalf = lane >> 4;
    const __bf16* base = src + (size_t)(col0 + n) * ld + kb + 8 * khalf;
    v8bf lo = *(const v8bf*)(base);
    v8bf hi = *(const v8bf*)(base + 16);
    v16bf b;
#pragma unroll
    for (int i = 0; i < 8; ++i) { b[i] = lo[i]; b[i + 8] = hi[i]; }
    return b;
}

// ---------------------------------------------------------------------------
// TDM: stage a 16x32 int tile of adj (row stride NN) into LDS at lds_off.
// D# per CDNA5 ISA 8.3/8.4: group0 = control/lds/global addr, group1 = dims/tiles/strides.
// This toolchain exposes the 6-arg builtin: (v4u, v8i, v4i, v4i, v8i, cpol).
__device__ inline void tdm_load_adj_tile(const int* __restrict__ gsrc, unsigned lds_off) {
    unsigned long long ga = (unsigned long long)(size_t)gsrc;
    v4u g0;
    g0[0] = 1u;                                        // count=1, user mode, no gather
    g0[1] = lds_off;                                   // lds_addr (bytes)
    g0[2] = (unsigned)(ga & 0xffffffffu);              // global_addr[31:0]
    g0[3] = (unsigned)((ga >> 32) & 0x01ffffffu)       // global_addr[56:32]
          | (2u << 30);                                // type = 2 ("image")
    v8i g1;
    g1[0] = 0x00020000;                                // data_size = 2 (4B); mask/flags = 0
    g1[1] = (int)((unsigned)NN << 16);                 // tensor_dim0[15:0]  (3072)
    g1[2] = (int)((unsigned)NN << 16);                 // tensor_dim0[31:16]=0 | tensor_dim1[15:0]
    g1[3] = (int)(32u << 16);                          // tensor_dim1[31:16]=0 | tile_dim0 = 32
    g1[4] = 16;                                        // tile_dim1 = 16, tile_dim2 = 0
    g1[5] = NN;                                        // tensor_dim0_stride[31:0] = 3072
    g1[6] = 0;                                         // stride0[47:32] | stride1[15:0]
    g1[7] = 0;                                         // tensor_dim1_stride[47:16]
    v4i g2 = {0, 0, 0, 0};
    v4i g3 = {0, 0, 0, 0};
    v8i g4 = {0, 0, 0, 0, 0, 0, 0, 0};
    __builtin_amdgcn_tensor_load_to_lds(g0, g1, g2, g3, g4, 0);
}

// ---------------------------------------------------------------------------
__global__ void gat_cvt_bf16(const float* __restrict__ src, __bf16* __restrict__ dst, int count) {
    int i = blockIdx.x * blockDim.x + threadIdx.x;
    if (i < count) dst[i] = (__bf16)src[i];
}

// GEMM1: C = x @ W^T, stored transposed hT[n][m] (bf16).
// One wave owns a 16x64 strip (1 m-tile x 4 n-tiles); K loop double-buffered so each
// WMMA group consumes fragments loaded one iteration earlier.
__global__ void gat_gemm_xw(const __bf16* __restrict__ xb, const __bf16* __restrict__ wb,
                            __bf16* __restrict__ hT) {
    const int wtile = blockIdx.x * 8 + (threadIdx.x >> 5);   // [0, 1536)
    const int ngrp  = wtile & 7;      // 8 groups of 4 n-tiles
    const int mtile = wtile >> 3;     // NN/16 = 192
    const int m0 = mtile * 16, n0 = ngrp * 64;

    v8f acc[4] = {};
    v16bf a = load_frag_a_rowmajor(xb, INF_, m0, 0);
    v16bf b[4];
#pragma unroll
    for (int dt = 0; dt < 4; ++dt) b[dt] = load_frag_b_kcontig(wb, (size_t)INF_, n0 + dt * 16, 0);

    for (int kb = 0; kb < INF_ - 32; kb += 32) {
        v16bf an = load_frag_a_rowmajor(xb, INF_, m0, kb + 32);
        v16bf bn[4];
#pragma unroll
        for (int dt = 0; dt < 4; ++dt)
            bn[dt] = load_frag_b_kcontig(wb, (size_t)INF_, n0 + dt * 16, kb + 32);
#pragma unroll
        for (int dt = 0; dt < 4; ++dt)
            acc[dt] = __builtin_amdgcn_wmma_f32_16x16x32_bf16(false, a, false, b[dt],
                                                              (short)0, acc[dt], false, false);
        a = an;
#pragma unroll
        for (int dt = 0; dt < 4; ++dt) b[dt] = bn[dt];
    }
#pragma unroll
    for (int dt = 0; dt < 4; ++dt)
        acc[dt] = __builtin_amdgcn_wmma_f32_16x16x32_bf16(false, a, false, b[dt],
                                                          (short)0, acc[dt], false, false);

    const int lane = threadIdx.x & 31;
    const int n = lane & 15, half = lane >> 4;
#pragma unroll
    for (int dt = 0; dt < 4; ++dt) {
#pragma unroll
        for (int r = 0; r < 8; ++r) {
            int m = m0 + r + 8 * half;             // C layout: VGPR r -> M = r (+8 hi lanes)
            hT[(size_t)(n0 + dt * 16 + n) * NN + m] = (__bf16)acc[dt][r];
        }
    }
}

// e_src[i,h] = sum_d h[i,h,d]*a_src[h,d] ; same for e_dst. One thread per (i,h).
__global__ void gat_edots(const __bf16* __restrict__ hT,
                          const float* __restrict__ a_src, const float* __restrict__ a_dst,
                          float* __restrict__ e_src, float* __restrict__ e_dst) {
    int t = blockIdx.x * blockDim.x + threadIdx.x;   // [0, NN*NHEAD)
    if (t >= NN * NHEAD) return;
    int i = t >> 3, hh = t & 7;
    float ss = 0.f, sd = 0.f;
#pragma unroll 8
    for (int d = 0; d < DHEAD; ++d) {
        float v = (float)hT[(size_t)(hh * DHEAD + d) * NN + i];
        ss = fmaf(v, a_src[hh * DHEAD + d], ss);
        sd = fmaf(v, a_dst[hh * DHEAD + d], sd);
    }
    e_src[i * NHEAD + hh] = ss;
    e_dst[i * NHEAD + hh] = sd;
}

// Pass A: per (row i, head h) masked online-softmax stats: row_max and 1/sum(exp).
__global__ void gat_rowstats(const int* __restrict__ adj,
                             const float* __restrict__ e_src, const float* __restrict__ e_dst,
                             float* __restrict__ row_max, float* __restrict__ row_rinv) {
    __shared__ float sm[256 * NHEAD];
    __shared__ float ssum[256 * NHEAD];
    const int i = blockIdx.x;
    const int tid = threadIdx.x;
    const float NEG = -1.0e30f;

    float es[NHEAD];
#pragma unroll
    for (int hh = 0; hh < NHEAD; ++hh) es[hh] = e_src[i * NHEAD + hh];

    float m[NHEAD], s[NHEAD];
#pragma unroll
    for (int hh = 0; hh < NHEAD; ++hh) { m[hh] = NEG; s[hh] = 0.f; }

    const int* adjRow = adj + (size_t)i * NN;
    for (int j = tid; j < NN; j += 256) {
        bool mask = (adjRow[j] != 0) || (j == i);
        if (!mask) continue;
#pragma unroll
        for (int hh = 0; hh < NHEAD; ++hh) {
            float v = es[hh] + e_dst[j * NHEAD + hh];
            v = v > 0.f ? v : 0.2f * v;            // leaky_relu(0.2)
            float nm = fmaxf(m[hh], v);
            s[hh] = s[hh] * __expf(m[hh] - nm) + __expf(v - nm);
            m[hh] = nm;
        }
    }
#pragma unroll
    for (int hh = 0; hh < NHEAD; ++hh) { sm[tid * NHEAD + hh] = m[hh]; ssum[tid * NHEAD + hh] = s[hh]; }
    __syncthreads();

    if (tid < NHEAD) {
        float M = NEG, S = 0.f;
        for (int t = 0; t < 256; ++t) {
            float mt = sm[t * NHEAD + tid], st = ssum[t * NHEAD + tid];
            float nm = fmaxf(M, mt);
            S = S * __expf(M - nm) + st * __expf(mt - nm);
            M = nm;
        }
        row_max[i * NHEAD + tid]  = M;
        row_rinv[i * NHEAD + tid] = 1.0f / S;
    }
}

// Pass B: fused aggregation.
// Block = one 16-row i-tile; wave w = head w, owning all 4 d-tiles (4 accumulators).
// Per 32-j chunk: wave 0 issues one TDM tensor_load_to_lds for the adj 16x32 tile and
// waits on TENSORcnt; the block stages the 1KB e_dst chunk cooperatively; each wave
// then builds its alpha A-fragment once from LDS (ds_load + v_exp_f32) and issues 4 WMMAs.
__global__ void gat_aggregate(const int* __restrict__ adj, const __bf16* __restrict__ hT,
                              const float* __restrict__ e_src, const float* __restrict__ e_dst,
                              const float* __restrict__ row_max, const float* __restrict__ row_rinv,
                              float* __restrict__ out) {
    __shared__ int   adjS[16 * 32];     // [m][jj]  (TDM destination)
    __shared__ float edS[32 * NHEAD];   // [jj][hh] (matches e_dst chunk layout)

    const int itile = blockIdx.x;       // [0, 192)
    const int i0    = itile * 16;
    const int hh    = threadIdx.x >> 5; // wave id = head
    const int lane  = threadIdx.x & 31;
    const int mrow  = lane & 15;
    const int half  = lane >> 4;
    const int iRow  = i0 + mrow;
    const int tid   = threadIdx.x;

    const float s_i  = e_src[iRow * NHEAD + hh];
    const float rmax = row_max[iRow * NHEAD + hh];
    const float rinv = row_rinv[iRow * NHEAD + hh];
    const unsigned adjS_lds = (unsigned)(size_t)(void*)adjS;   // LDS byte offset (low 32 bits)

    // A-fragment element -> K offset (ISA 7.12.2, 16-bit A 16x32)
    int koff[16];
#pragma unroll
    for (int idx = 0; idx < 16; ++idx) {
        int v = idx >> 1, p = idx & 1;
        koff[idx] = ((v & 3) * 2 + p) + (v >= 4 ? 16 : 0) + 8 * half;
    }

    v8f acc[4] = {};
    for (int j0 = 0; j0 < NN; j0 += 32) {
        __syncthreads();   // previous chunk's LDS reads done before overwrite
        // TDM DMA: one descriptor per block moves the 16x32 adj tile into LDS.
        if (hh == 0) {
            tdm_load_adj_tile(adj + (size_t)i0 * NN + j0, adjS_lds);
            __builtin_amdgcn_s_wait_tensorcnt(0);
        }
        // Stage e_dst tile: 256 contiguous floats.
        edS[tid] = e_dst[(size_t)j0 * NHEAD + tid];
        __syncthreads();

        // Touch next e_dst chunk (global_prefetch_b8).
        if (j0 + 32 < NN) {
            __builtin_prefetch(&e_dst[(size_t)(j0 + 32) * NHEAD + tid], 0, 1);
        }

        // B fragments first (long-latency global), alpha build overlaps them.
        v16bf b[4];
#pragma unroll
        for (int dt = 0; dt < 4; ++dt)
            b[dt] = load_frag_b_kcontig(hT, (size_t)NN, hh * DHEAD + dt * 16, j0);

        // Build alpha fragment once for this (i-tile, head) and j-chunk.
        v16bf a;
#pragma unroll
        for (int idx = 0; idx < 16; ++idx) {
            int k = koff[idx];
            int j = j0 + k;
            bool mask = (adjS[mrow * 32 + k] != 0) || (j == iRow);
            float v = s_i + edS[k * NHEAD + hh];
            v = v > 0.f ? v : 0.2f * v;
            float alpha = mask ? __expf(v - rmax) * rinv : 0.f;
            a[idx] = (__bf16)alpha;
        }

#pragma unroll
        for (int dt = 0; dt < 4; ++dt)
            acc[dt] = __builtin_amdgcn_wmma_f32_16x16x32_bf16(false, a, false, b[dt],
                                                              (short)0, acc[dt], false, false);
    }

#pragma unroll
    for (int dt = 0; dt < 4; ++dt) {
#pragma unroll
        for (int r = 0; r < 8; ++r) {
            int row = i0 + r + 8 * half;
            int col = hh * DHEAD + dt * 16 + mrow;
            out[(size_t)row * HD + col] = acc[dt][r];
        }
    }
}

// ---------------------------------------------------------------------------
extern "C" void kernel_launch(void* const* d_in, const int* in_sizes, int n_in,
                              void* d_out, int out_size, void* d_ws, size_t ws_size,
                              hipStream_t stream) {
    const float* x     = (const float*)d_in[0];
    const int*   adj   = (const int*)  d_in[1];
    const float* W     = (const float*)d_in[2];
    const float* a_src = (const float*)d_in[3];
    const float* a_dst = (const float*)d_in[4];
    float* out = (float*)d_out;

    char* ws = (char*)d_ws;
    size_t off = 0;
    __bf16* x_bf = (__bf16*)(ws + off); off += (size_t)NN * INF_ * sizeof(__bf16);
    __bf16* w_bf = (__bf16*)(ws + off); off += (size_t)HD * INF_ * sizeof(__bf16);
    __bf16* hT   = (__bf16*)(ws + off); off += (size_t)HD * NN   * sizeof(__bf16);
    float* e_src = (float*)(ws + off);  off += (size_t)NN * NHEAD * sizeof(float);
    float* e_dst = (float*)(ws + off);  off += (size_t)NN * NHEAD * sizeof(float);
    float* rmax  = (float*)(ws + off);  off += (size_t)NN * NHEAD * sizeof(float);
    float* rinv  = (float*)(ws + off);  off += (size_t)NN * NHEAD * sizeof(float);
    (void)ws_size; (void)in_sizes; (void)n_in; (void)out_size;

    gat_cvt_bf16<<<(NN * INF_ + 255) / 256, 256, 0, stream>>>(x, x_bf, NN * INF_);
    gat_cvt_bf16<<<(HD * INF_ + 255) / 256, 256, 0, stream>>>(W, w_bf, HD * INF_);

    // 192 m-tiles * 8 n-groups = 1536 wave-strips / 8 waves per block
    gat_gemm_xw<<<192, 256, 0, stream>>>(x_bf, w_bf, hT);

    gat_edots<<<(NN * NHEAD + 255) / 256, 256, 0, stream>>>(hT, a_src, a_dst, e_src, e_dst);

    gat_rowstats<<<NN, 256, 0, stream>>>(adj, e_src, e_dst, rmax, rinv);

    // one block per 16-row i-tile; wave = head, 4 d-tiles per wave
    gat_aggregate<<<NN / 16, 256, 0, stream>>>(adj, hT, e_src, e_dst, rmax, rinv, out);
}
